// JointHistParzenTorch_13529146982827
// MI455X (gfx1250) — compile-verified, matrix operand order checked by reading, not run on hardware
//
#include <hip/hip_runtime.h>

#define NUM_BINS 64
#define NIMG     16
#define NPIX     65536          // 256*256 pixels per image
#define CHUNKS   64             // pixel splits per image
#define PC       (NPIX / CHUNKS) // 1024 pixels per wave
#define TILE_P   256            // pixels staged in LDS per outer step

typedef __attribute__((ext_vector_type(16))) _Float16 v16h;
typedef __attribute__((ext_vector_type(8)))  float    v8f;

// Order-preserving float <-> uint mapping (for atomic min/max on uints)
__device__ __forceinline__ unsigned fmono(float f) {
  unsigned u = __float_as_uint(f);
  return (u & 0x80000000u) ? ~u : (u | 0x80000000u);
}
__device__ __forceinline__ float fmono_inv(unsigned u) {
  return __uint_as_float((u & 0x80000000u) ? (u & 0x7fffffffu) : ~u);
}

// cubic B-spline, BRANCHLESS closed form:
//   B3(x) = ( max(2-|x|,0)^3 - 4*max(1-|x|,0)^3 ) / 6
// identical to the piecewise definition on all intervals; pure VALU.
__device__ __forceinline__ float bspline3(float x) {
  float t  = fabsf(x);
  float u  = fmaxf(2.0f - t, 0.0f);
  float v  = fmaxf(1.0f - t, 0.0f);
  float u3 = u * u * u;
  float v3 = v * v * v;
  return 0.16666667f * u3 - 0.66666667f * v3;
}

__global__ void init_kernel(float* __restrict__ out, int n, unsigned* __restrict__ ws) {
  for (int i = blockIdx.x * blockDim.x + threadIdx.x; i < n; i += gridDim.x * blockDim.x)
    out[i] = 0.0f;
  if (blockIdx.x == 0 && threadIdx.x < 4)
    ws[threadIdx.x] = (threadIdx.x & 1) ? 0u : 0xFFFFFFFFu; // [min,max,min,max]
}

__global__ void minmax_kernel(const float* __restrict__ r, const float* __restrict__ t,
                              unsigned* __restrict__ ws, int n) {
  __shared__ float s[4][256];
  float rmin = 3.4e38f, rmax = -3.4e38f, tmin = 3.4e38f, tmax = -3.4e38f;
  for (int i = blockIdx.x * blockDim.x + threadIdx.x; i < n; i += gridDim.x * blockDim.x) {
    float a = r[i]; rmin = fminf(rmin, a); rmax = fmaxf(rmax, a);
    float b = t[i]; tmin = fminf(tmin, b); tmax = fmaxf(tmax, b);
  }
  int tid = threadIdx.x;
  s[0][tid] = rmin; s[1][tid] = rmax; s[2][tid] = tmin; s[3][tid] = tmax;
  __syncthreads();
  for (int off = 128; off > 0; off >>= 1) {
    if (tid < off) {
      s[0][tid] = fminf(s[0][tid], s[0][tid + off]);
      s[1][tid] = fmaxf(s[1][tid], s[1][tid + off]);
      s[2][tid] = fminf(s[2][tid], s[2][tid + off]);
      s[3][tid] = fmaxf(s[3][tid], s[3][tid + off]);
    }
    __syncthreads();
  }
  if (tid == 0) {
    atomicMin(&ws[0], fmono(s[0][0]));
    atomicMax(&ws[1], fmono(s[1][0]));
    atomicMin(&ws[2], fmono(s[2][0]));
    atomicMax(&ws[3], fmono(s[3][0]));
  }
}

// One wave32 per block. Each wave computes the full 64x64 joint histogram
// (4x4 grid of 16x16 WMMA tiles) over its pixel chunk, then atomically
// accumulates into the per-batch output.
__global__ void __launch_bounds__(32)
hist_wmma_kernel(const float* __restrict__ img_ref, const float* __restrict__ img_tar,
                 const float* __restrict__ bins_ref, const float* __restrict__ bins_tar,
                 float* __restrict__ out, const unsigned* __restrict__ ws) {
  const int n     = blockIdx.x;   // batch
  const int chunk = blockIdx.y;   // pixel chunk
  const int lane  = threadIdx.x;  // 0..31
  const int half  = lane >> 4;    // 0 or 1 (K-range half per WMMA layout)
  const int mc    = lane & 15;    // row (A) / col (B) within 16x16 tile

  const float rmin = fmono_inv(ws[0]), rmax = fmono_inv(ws[1]);
  const float tmin = fmono_inv(ws[2]), tmax = fmono_inv(ws[3]);
  const float rs = 64.0f / fmaxf(rmax - rmin, 1e-20f); // bin-units scale
  const float ts = 64.0f / fmaxf(tmax - tmin, 1e-20f);

  // bin coordinates in units of EPS: bins[m]/EPS
  float binA[4], binB[4];
#pragma unroll
  for (int ti = 0; ti < 4; ++ti) {
    binA[ti] = bins_ref[mc + 16 * ti] * 64.0f;
    binB[ti] = bins_tar[mc + 16 * ti] * 64.0f;
  }

  __shared__ float sr[TILE_P];
  __shared__ float st[TILE_P];

  const float* rp = img_ref + (size_t)n * NPIX + (size_t)chunk * PC;
  const float* tp = img_tar + (size_t)n * NPIX + (size_t)chunk * PC;

  v8f acc[4][4] = {};

#pragma unroll 1
  for (int outer = 0; outer < PC; outer += TILE_P) {
    __syncthreads(); // protect LDS reuse
#pragma unroll
    for (int i = lane; i < TILE_P; i += 32) {
      sr[i] = (rp[outer + i] - rmin) * rs; // normalized pixel in [0,64]
      st[i] = (tp[outer + i] - tmin) * ts;
    }
    __syncthreads();

#pragma unroll 1
    for (int kk = 0; kk < TILE_P; kk += 32) {
      v16h a[4], b[4];
      // 16-bit A-matrix layout: lane<16 holds K 0..7 (elems 0-7), 16..23 (8-15);
      // lane>=16 holds K 8..15, 24..31. B symmetric with N=lane&15.
#pragma unroll
      for (int j = 0; j < 16; ++j) {
        int k = kk + half * 8 + (j < 8 ? j : j + 8);
        float vr = sr[k];
        float vt = st[k];
#pragma unroll
        for (int ti = 0; ti < 4; ++ti) {
          a[ti][j] = (_Float16)bspline3(binA[ti] - vr);
          b[ti][j] = (_Float16)bspline3(binB[ti] - vt);
        }
      }
#pragma unroll
      for (int ti = 0; ti < 4; ++ti)
#pragma unroll
        for (int tj = 0; tj < 4; ++tj)
          acc[ti][tj] = __builtin_amdgcn_wmma_f32_16x16x32_f16(
              false, a[ti], false, b[tj], (short)0, acc[ti][tj], false, false);
    }
  }

  // C/D layout: VGPR v, lane<16 -> row v, col lane; lane>=16 -> row v+8, col lane-16
  float* outn = out + (size_t)n * NUM_BINS * NUM_BINS;
#pragma unroll
  for (int ti = 0; ti < 4; ++ti)
#pragma unroll
    for (int tj = 0; tj < 4; ++tj)
#pragma unroll
      for (int v = 0; v < 8; ++v) {
        int row = ti * 16 + v + 8 * half;
        int col = tj * 16 + mc;
        atomicAdd(&outn[row * NUM_BINS + col], acc[ti][tj][v] * 4096.0f); // 1/EPS^2
      }
}

extern "C" void kernel_launch(void* const* d_in, const int* in_sizes, int n_in,
                              void* d_out, int out_size, void* d_ws, size_t ws_size,
                              hipStream_t stream) {
  const float* img_ref  = (const float*)d_in[0];
  const float* img_tar  = (const float*)d_in[1];
  const float* bins_ref = (const float*)d_in[2];
  const float* bins_tar = (const float*)d_in[3];
  float*    out = (float*)d_out;
  unsigned* ws  = (unsigned*)d_ws;

  (void)in_sizes; (void)n_in; (void)ws_size;

  init_kernel<<<256, 256, 0, stream>>>(out, out_size, ws);
  minmax_kernel<<<512, 256, 0, stream>>>(img_ref, img_tar, ws, NIMG * NPIX);

  dim3 grid(NIMG, CHUNKS);
  hist_wmma_kernel<<<grid, 32, 0, stream>>>(img_ref, img_tar, bins_ref, bins_tar, out, ws);
}
